// SelfAttention_v1_58067957842421
// MI455X (gfx1250) — matrix operand order repaired and checked by reading, not compiled, not run on hardware
//
#include <hip/hip_runtime.h>

#define SEQ 4096
#define DIM 1024

typedef __bf16 bf16;
typedef __attribute__((ext_vector_type(8)))  __bf16 v8bf;
typedef __attribute__((ext_vector_type(16))) __bf16 v16bf;
typedef __attribute__((ext_vector_type(8)))  float  v8f;
typedef __attribute__((ext_vector_type(4)))  unsigned int v4u;
typedef __attribute__((ext_vector_type(8)))  int v8i;
typedef __attribute__((ext_vector_type(4)))  int v4i;

#define BM 128
#define BN 128
#define BK 32
#define LDSP 48                       // 32 data + 16 pad elems -> 96B stride (16B aligned)
#define TILE_BYTES (BM * LDSP * 2)    // 12288 B per buffered tile

__device__ __forceinline__ v16bf cat8(v8bf lo, v8bf hi) {
  return __builtin_shufflevector(lo, hi, 0,1,2,3,4,5,6,7,8,9,10,11,12,13,14,15);
}

// ---------------------------------------------------------------------------
// Tensor Data Mover: DMA one 128(rows) x 32(bf16) tile, row stride = K elems,
// into LDS at lds_off, inserting 32B padding after every 64B row so the LDS
// layout is [128][48] bf16. D# per CDNA5 ISA:
//   g1[0]: data_size=1(2B) | pad_enable | pad_interval=3(16 dw) | pad_amount=7(8 dw)
// ---------------------------------------------------------------------------
__device__ __forceinline__ void tdm_load_tile_128x32(unsigned lds_off,
                                                     const bf16* gsrc,
                                                     unsigned rowStrideElems)
{
  unsigned long long ga = (unsigned long long)(size_t)gsrc;
  v4u g0;
  g0.x = 1u;                                                  // count=1, user desc
  g0.y = lds_off;                                             // lds_addr (bytes)
  g0.z = (unsigned)ga;                                        // global_addr[31:0]
  g0.w = (unsigned)((ga >> 32) & 0x01FFFFFFu) | (2u << 30);   // addr[56:32] | type=2
  v8i g1;
  g1[0] = (int)0x0ED10000u;      // data_size=2B, pad_enable, interval=16dw, amount=8dw
  g1[1] = (int)(BK << 16);       // tensor_dim0 = 32 (tile-relative width)
  g1[2] = (int)(BM << 16);       // tensor_dim1 = 128
  g1[3] = (int)(BK << 16);       // tile_dim0 = 32
  g1[4] = (int)BM;               // tile_dim1 = 128, tile_dim2 = 0
  g1[5] = (int)rowStrideElems;   // tensor_dim0_stride[31:0]
  g1[6] = 0;
  g1[7] = 0;
  v4i gz = {0, 0, 0, 0};
#if defined(__clang_major__) && (__clang_major__ >= 23)
  v8i gz8 = {0, 0, 0, 0, 0, 0, 0, 0};
  __builtin_amdgcn_tensor_load_to_lds(g0, g1, gz, gz, gz8, 0);
#else
  __builtin_amdgcn_tensor_load_to_lds(g0, g1, gz, gz, 0);
#endif
}

// ---------------------------------------------------------------------------
// x: f32 -> bf16 straight copy
// ---------------------------------------------------------------------------
__global__ __launch_bounds__(256)
void convert_x_kernel(const float* __restrict__ x, bf16* __restrict__ xb)
{
  size_t i = (size_t)blockIdx.x * 256 + threadIdx.x;
  if (i < (size_t)SEQ * DIM) xb[i] = (bf16)x[i];
}

// ---------------------------------------------------------------------------
// f32 [R x C] -> bf16 [C x R] transpose (LDS tiled), with scale fold
// ---------------------------------------------------------------------------
__global__ __launch_bounds__(256)
void transpose_conv_kernel(const float* __restrict__ src, bf16* __restrict__ dst,
                           int R, int C, float scale)
{
  __shared__ float t[32][33];
  const int bx = blockIdx.x * 32;          // col base in src
  const int by = blockIdx.y * 32;          // row base in src
  const int tx = threadIdx.x & 31;
  const int ty = threadIdx.x >> 5;         // 0..7
  #pragma unroll
  for (int i = ty; i < 32; i += 8)
    t[i][tx] = src[(size_t)(by + i) * C + bx + tx] * scale;
  __syncthreads();
  #pragma unroll
  for (int i = ty; i < 32; i += 8)
    dst[(size_t)(bx + i) * R + by + tx] = (bf16)t[tx][i];
}

// ---------------------------------------------------------------------------
// bf16 [R x C] -> bf16 [C x R] transpose (for V)
// ---------------------------------------------------------------------------
__global__ __launch_bounds__(256)
void transpose_bf16_kernel(const bf16* __restrict__ src, bf16* __restrict__ dst,
                           int R, int C)
{
  __shared__ bf16 t[32][33];
  const int bx = blockIdx.x * 32;
  const int by = blockIdx.y * 32;
  const int tx = threadIdx.x & 31;
  const int ty = threadIdx.x >> 5;
  #pragma unroll
  for (int i = ty; i < 32; i += 8)
    t[i][tx] = src[(size_t)(by + i) * C + bx + tx];
  __syncthreads();
  #pragma unroll
  for (int i = ty; i < 32; i += 8)
    dst[(size_t)(bx + i) * R + by + tx] = t[tx][i];
}

// ---------------------------------------------------------------------------
// NT GEMM on v_wmma_f32_16x16x32_bf16 with TDM-fed double-buffered LDS.
//   C[M x N] = A[M x K] * B^T, A row-major [M x K], B row-major [N x K]
//   (both operands k-contiguous -> all LDS traffic is b128)
// Block tile 128x128, K-step 32, 4 waves (2x2), wave tile 64x64 = 4x4 frags.
// Wave 0 drives the TDM: while all waves compute on buffer b, the TDM fills
// buffer b^1; completion is published via s_wait_tensorcnt + barrier.
// ---------------------------------------------------------------------------
template<bool OUT_F32>
__global__ __launch_bounds__(128)
void gemm_nt_wmma(const bf16* __restrict__ A, const bf16* __restrict__ B,
                  void* __restrict__ Cout, int M, int N, int K)
{
  __shared__ bf16 As[2][BM][LDSP];
  __shared__ bf16 Bs[2][BN][LDSP];

  const int tid    = threadIdx.x;
  const int lane   = tid & 31;
  const int wave   = tid >> 5;       // 0..3
  const int waveM  = wave >> 1;      // 0..1
  const int waveN  = wave & 1;       // 0..1
  const int laneLo = lane & 15;
  const int laneHi = lane >> 4;

  const int rowBase = blockIdx.y * BM;
  const int colBase = blockIdx.x * BN;

  const unsigned asBase = (unsigned)(size_t)&As[0][0][0];
  const unsigned bsBase = (unsigned)(size_t)&Bs[0][0][0];

  const v8f vzero = {0.f,0.f,0.f,0.f,0.f,0.f,0.f,0.f};
  v8f acc[4][4];
  #pragma unroll
  for (int mi = 0; mi < 4; ++mi)
    #pragma unroll
    for (int ni = 0; ni < 4; ++ni) acc[mi][ni] = vzero;

  if (wave == 0) {   // prime buffer 0
    tdm_load_tile_128x32(asBase, A + (size_t)rowBase * K, (unsigned)K);
    tdm_load_tile_128x32(bsBase, B + (size_t)colBase * K, (unsigned)K);
  }

  int buf = 0;
  for (int k0 = 0; k0 < K; k0 += BK) {
    if (wave == 0) __builtin_amdgcn_s_wait_tensorcnt(0);
    __syncthreads();                       // tile `buf` visible to all waves

    if (wave == 0 && (k0 + BK) < K) {      // TDM fills buf^1 during compute
      tdm_load_tile_128x32(asBase + (unsigned)(buf ^ 1) * TILE_BYTES,
                           A + (size_t)rowBase * K + k0 + BK, (unsigned)K);
      tdm_load_tile_128x32(bsBase + (unsigned)(buf ^ 1) * TILE_BYTES,
                           B + (size_t)colBase * K + k0 + BK, (unsigned)K);
    }

    // fragment loads (ISA 7.12.2 wave32 layouts), all ds_load_b128
    v16bf aF[4], bF[4];
    #pragma unroll
    for (int mi = 0; mi < 4; ++mi) {
      const int r  = waveM * 64 + mi * 16 + laneLo;
      const int kb = laneHi * 8;
      aF[mi] = cat8(*(const v8bf*)&As[buf][r][kb],
                    *(const v8bf*)&As[buf][r][kb + 16]);
    }
    #pragma unroll
    for (int ni = 0; ni < 4; ++ni) {
      const int c  = waveN * 64 + ni * 16 + laneLo;
      const int kb = laneHi * 8;
      bF[ni] = cat8(*(const v8bf*)&Bs[buf][c][kb],
                    *(const v8bf*)&Bs[buf][c][kb + 16]);
    }

    #pragma unroll
    for (int mi = 0; mi < 4; ++mi)
      #pragma unroll
      for (int ni = 0; ni < 4; ++ni)
        acc[mi][ni] = __builtin_amdgcn_wmma_f32_16x16x32_bf16(
            false, aF[mi], false, bF[ni], (short)0, acc[mi][ni], false, false);

    __syncthreads();                       // done reading `buf`
    buf ^= 1;
  }

  // epilogue: C/D layout -> lane<16: M=vgpr, lane>=16: M=vgpr+8
  #pragma unroll
  for (int mi = 0; mi < 4; ++mi) {
    const int row0 = rowBase + waveM * 64 + mi * 16 + laneHi * 8;
    #pragma unroll
    for (int ni = 0; ni < 4; ++ni) {
      const int col = colBase + waveN * 64 + ni * 16 + laneLo;
      #pragma unroll
      for (int r = 0; r < 8; ++r) {
        if constexpr (OUT_F32)
          ((float*)Cout)[(size_t)(row0 + r) * N + col] = acc[mi][ni][r];
        else
          ((bf16*)Cout)[(size_t)(row0 + r) * N + col] = (bf16)acc[mi][ni][r];
      }
    }
  }
}

// ---------------------------------------------------------------------------
// Row softmax over 4096x4096 bf16 scores, in place, f32 math, b128 loads.
// ---------------------------------------------------------------------------
__global__ __launch_bounds__(256)
void softmax_rows(bf16* __restrict__ S)
{
  __shared__ float red[256];
  const int tid = threadIdx.x;
  v8bf* p = (v8bf*)(S + (size_t)blockIdx.x * SEQ);
  const int NV = SEQ / 8;   // 512 vectors per row

  float mx = -3.4e38f;
  for (int i = tid; i < NV; i += 256) {
    v8bf v = p[i];
    #pragma unroll
    for (int j = 0; j < 8; ++j) mx = fmaxf(mx, (float)v[j]);
  }
  red[tid] = mx; __syncthreads();
  for (int s = 128; s > 0; s >>= 1) {
    if (tid < s) red[tid] = fmaxf(red[tid], red[tid + s]);
    __syncthreads();
  }
  mx = red[0]; __syncthreads();

  float sum = 0.f;
  for (int i = tid; i < NV; i += 256) {
    v8bf v = p[i];
    #pragma unroll
    for (int j = 0; j < 8; ++j) sum += __expf((float)v[j] - mx);
  }
  red[tid] = sum; __syncthreads();
  for (int s = 128; s > 0; s >>= 1) {
    if (tid < s) red[tid] += red[tid + s];
    __syncthreads();
  }
  const float inv = 1.0f / red[0];
  __syncthreads();

  for (int i = tid; i < NV; i += 256) {
    v8bf v = p[i], o;
    #pragma unroll
    for (int j = 0; j < 8; ++j) o[j] = (bf16)(__expf((float)v[j] - mx) * inv);
    p[i] = o;
  }
}

// ---------------------------------------------------------------------------
extern "C" void kernel_launch(void* const* d_in, const int* in_sizes, int n_in,
                              void* d_out, int out_size, void* d_ws, size_t ws_size,
                              hipStream_t stream)
{
  const float* x  = (const float*)d_in[0];
  const float* Wq = (const float*)d_in[1];
  const float* Wk = (const float*)d_in[2];
  const float* Wv = (const float*)d_in[3];
  float* out = (float*)d_out;

  // ws (bf16): x | Wq^T | Wk^T | Wv^T | Q | K | V | V^T | S   (~78 MB)
  bf16* Xb  = (bf16*)d_ws;
  bf16* Wqt = Xb  + (size_t)SEQ * DIM;
  bf16* Wkt = Wqt + (size_t)DIM * DIM;
  bf16* Wvt = Wkt + (size_t)DIM * DIM;
  bf16* Qb  = Wvt + (size_t)DIM * DIM;
  bf16* Kb  = Qb  + (size_t)SEQ * DIM;
  bf16* Vb  = Kb  + (size_t)SEQ * DIM;
  bf16* Vt  = Vb  + (size_t)SEQ * DIM;   // [DIM x SEQ]
  bf16* Sb  = Vt  + (size_t)DIM * SEQ;   // [SEQ x SEQ]

  convert_x_kernel<<<(SEQ * DIM) / 256, 256, 0, stream>>>(x, Xb);

  dim3 gt(DIM / 32, DIM / 32);   // weight transposes (scale folded into Wq)
  transpose_conv_kernel<<<gt, 256, 0, stream>>>(Wq, Wqt, DIM, DIM, 0.03125f);
  transpose_conv_kernel<<<gt, 256, 0, stream>>>(Wk, Wkt, DIM, DIM, 1.0f);
  transpose_conv_kernel<<<gt, 256, 0, stream>>>(Wv, Wvt, DIM, DIM, 1.0f);

  dim3 gp(DIM / BN, SEQ / BM);   // 8 x 32
  gemm_nt_wmma<false><<<gp, 128, 0, stream>>>(Xb, Wqt, Qb, SEQ, DIM, DIM);
  gemm_nt_wmma<false><<<gp, 128, 0, stream>>>(Xb, Wkt, Kb, SEQ, DIM, DIM);
  gemm_nt_wmma<false><<<gp, 128, 0, stream>>>(Xb, Wvt, Vb, SEQ, DIM, DIM);

  dim3 gs(SEQ / BN, SEQ / BM);   // 32 x 32: S = Q * K^T (K already [N x K] form)
  gemm_nt_wmma<false><<<gs, 128, 0, stream>>>(Qb, Kb, Sb, SEQ, SEQ, DIM);

  softmax_rows<<<SEQ, 256, 0, stream>>>(Sb);

  dim3 gv(DIM / 32, SEQ / 32);   // V [SEQ x DIM] -> V^T [DIM x SEQ]
  transpose_bf16_kernel<<<gv, 256, 0, stream>>>(Vb, Vt, SEQ, DIM);

  gemm_nt_wmma<true><<<gp, 128, 0, stream>>>(Sb, Vt, out, SEQ, DIM, SEQ);
}
